// MultiheadSelfAttention_31516470018496
// MI455X (gfx1250) — matrix operand order repaired and checked by reading, not compiled, not run on hardware
//
#include <hip/hip_runtime.h>
#include <hip/hip_bf16.h>

typedef __bf16 bf16_t;
typedef __attribute__((ext_vector_type(16))) __bf16 v16bf;
typedef __attribute__((ext_vector_type(8)))  float  v8f;

#define T_SEQ 1536
#define HID   2880
#define NH_   64
#define NKV_  8
#define DH    64
#define NQ    (NH_*DH)        // 4096
#define NKVD  (NKV_*DH)       // 512
#define NQKV  (NQ + 2*NKVD)   // 5120
#define TP    1664            // padded T for V^T (>= 1392+160)
#define WINDOW 128

union ABu { v16bf v; uint4 q[2]; };

// A fragment 16x32 bf16, row-major source (stride ld):
// lane<16: M=lane, elems=K[k0..k0+15]; lane>=16: M=lane-16, elems=K[k0+16..k0+31]
static __device__ inline v16bf wld_a(const bf16_t* base, int ld, int m0, int k0, int lane){
  int m    = m0 + (lane & 15);
  int half = lane >> 4;
  const bf16_t* p = base + (size_t)m * ld + k0 + half*16;
  ABu u;
  u.q[0] = *reinterpret_cast<const uint4*>(p);
  u.q[1] = *reinterpret_cast<const uint4*>(p + 8);
  return u.v;
}
// B fragment 32x16 bf16 from transposed source Bt[N][K] (stride ld):
// lane<16: N=lane, elems=K[k0..k0+15]; lane>=16: N=lane-16, elems=K[k0+16..k0+31]
static __device__ inline v16bf wld_b(const bf16_t* bt, int ld, int n0, int k0, int lane){
  int n = n0 + (lane & 15);
  const bf16_t* p = bt + (size_t)n * ld + k0 + (lane >> 4)*16;
  ABu u;
  u.q[0] = *reinterpret_cast<const uint4*>(p);
  u.q[1] = *reinterpret_cast<const uint4*>(p + 8);
  return u.v;
}
static __device__ inline v8f wmma_bf16(v16bf a, v16bf b, v8f c){
  return __builtin_amdgcn_wmma_f32_16x16x32_bf16(false, a, false, b, (short)0, c, false, false);
}

// ---------------- elementwise / layout kernels ----------------

__global__ void f32_to_bf16_kernel(const float* __restrict__ in, bf16_t* __restrict__ out, int n){
  int i = blockIdx.x*blockDim.x + threadIdx.x;
  if (i < n) out[i] = (bf16_t)in[i];
}

__global__ void build_wqkvT_kernel(const float* __restrict__ Wq, const float* __restrict__ Wk,
                                   const float* __restrict__ Wv, bf16_t* __restrict__ wt){
  int idx = blockIdx.x*blockDim.x + threadIdx.x;
  if (idx >= NQKV*HID) return;
  int nrow = idx / HID;      // output channel
  int k    = idx % HID;      // input channel
  float v;
  if      (nrow < NQ)        v = Wq[(size_t)k*NQ   + nrow];
  else if (nrow < NQ+NKVD)   v = Wk[(size_t)k*NKVD + (nrow - NQ)];
  else                       v = Wv[(size_t)k*NKVD + (nrow - NQ - NKVD)];
  wt[idx] = (bf16_t)v;
}

__global__ void build_woT_kernel(const float* __restrict__ Wo, bf16_t* __restrict__ wt){
  int idx = blockIdx.x*blockDim.x + threadIdx.x;
  if (idx >= HID*NQ) return;
  int n = idx / NQ;
  int k = idx % NQ;
  wt[idx] = (bf16_t)Wo[(size_t)k*HID + n];
}

__global__ void concat_bias_kernel(const float* __restrict__ bq, const float* __restrict__ bk,
                                   const float* __restrict__ bv, float* __restrict__ out){
  int i = blockIdx.x*blockDim.x + threadIdx.x;
  if (i >= NQKV) return;
  out[i] = (i < NQ) ? bq[i] : (i < NQ+NKVD) ? bk[i-NQ] : bv[i-NQ-NKVD];
}

// RoPE: inv_eff(i) = theta^(-2i/64) * (0.5/1 + 0.5/32) ; ang = (t/32)*inv_eff
// out[2i] = e*cos - o*sin ; out[2i+1] = e*sin + o*cos
#define L2THETA_OVER32 0.53733131f   // log2(150000)/32

__global__ void rope_q_kernel(const float* __restrict__ qkv, bf16_t* __restrict__ qb){
  int idx = blockIdx.x*blockDim.x + threadIdx.x;
  if (idx >= T_SEQ*(NQ/2)) return;
  int t   = idx / (NQ/2);
  int rem = idx % (NQ/2);
  int h = rem >> 5;
  int i = rem & 31;
  const float* row = qkv + (size_t)t*NQKV + h*DH;
  float e = row[2*i], o = row[2*i+1];
  float ang = ((float)t * (1.0f/32.0f)) * 0.515625f * exp2f(-(float)i * L2THETA_OVER32);
  float sn = __sinf(ang), cs = __cosf(ang);
  bf16_t* orow = qb + (size_t)t*NQ + h*DH;
  orow[2*i]   = (bf16_t)(e*cs - o*sn);
  orow[2*i+1] = (bf16_t)(e*sn + o*cs);
}

__global__ void rope_k_kernel(const float* __restrict__ qkv, bf16_t* __restrict__ kb){
  int idx = blockIdx.x*blockDim.x + threadIdx.x;
  if (idx >= T_SEQ*(NKVD/2)) return;
  int t   = idx / (NKVD/2);
  int rem = idx % (NKVD/2);
  int h = rem >> 5;
  int i = rem & 31;
  const float* row = qkv + (size_t)t*NQKV + NQ + h*DH;
  float e = row[2*i], o = row[2*i+1];
  float ang = ((float)t * (1.0f/32.0f)) * 0.515625f * exp2f(-(float)i * L2THETA_OVER32);
  float sn = __sinf(ang), cs = __cosf(ang);
  bf16_t* orow = kb + (size_t)t*NKVD + h*DH;
  orow[2*i]   = (bf16_t)(e*cs - o*sn);
  orow[2*i+1] = (bf16_t)(e*sn + o*cs);
}

// V^T: vT[(kvh*64+d)*TP + s] = v[s][kvh*64+d], zero pad s in [T_SEQ,TP)
__global__ void vtrans_kernel(const float* __restrict__ qkv, bf16_t* __restrict__ vT){
  int idx = blockIdx.x*blockDim.x + threadIdx.x;
  if (idx >= NKV_*DH*TP) return;
  int s  = idx % TP;
  int dk = idx / TP;
  float val = 0.0f;
  if (s < T_SEQ) val = qkv[(size_t)s*NQKV + NQ + NKVD + dk];
  vT[idx] = (bf16_t)val;
}

// ---------------- GEMM: C[M,N] f32 = A_bf16[M,K] * Bt_bf16[N,K]^T + bias ----------------
// block = 128 threads (4 waves) stacked in M; block tile 256x64, wave tile 64x64.
// All 4 waves share identical B fragments per k-step -> WGP$/L1 reuse.
// 16 WMMA per 16 b128 loads per lane: ~32 FLOP per cached byte.

__global__ __launch_bounds__(128) void gemm_bf16_kernel(const bf16_t* __restrict__ A,
    const bf16_t* __restrict__ Bt, const float* __restrict__ bias,
    float* __restrict__ C, int M, int N, int K){
  const int lane = threadIdx.x & 31;
  const int w    = threadIdx.x >> 5;
  const int m0 = blockIdx.y*256 + w*64;
  const int n0 = blockIdx.x*64;
  const v8f vzero = {};
  v8f c[4][4];
  #pragma unroll
  for (int i = 0; i < 4; ++i)
    #pragma unroll
    for (int j = 0; j < 4; ++j)
      c[i][j] = vzero;

  for (int k = 0; k < K; k += 32){
    v16bf a[4], b[4];
    #pragma unroll
    for (int i = 0; i < 4; ++i) a[i] = wld_a(A, K, m0 + i*16, k, lane);
    #pragma unroll
    for (int j = 0; j < 4; ++j) b[j] = wld_b(Bt, K, n0 + j*16, k, lane);
    #pragma unroll
    for (int i = 0; i < 4; ++i)
      #pragma unroll
      for (int j = 0; j < 4; ++j)
        c[i][j] = wmma_bf16(a[i], b[j], c[i][j]);
  }

  const int nc = lane & 15, hi = lane >> 4;
  #pragma unroll
  for (int j = 0; j < 4; ++j){
    float bv = bias[n0 + j*16 + nc];
    #pragma unroll
    for (int i = 0; i < 4; ++i){
      #pragma unroll
      for (int r = 0; r < 8; ++r){
        C[(size_t)(m0 + i*16 + 8*hi + r)*N + n0 + j*16 + nc] = c[i][j][r] + bv;
      }
    }
  }
}

// ---------------- attention: 1 wave per (16-query tile, head) ----------------
// sliding-window causal: q at t attends to s in [t-128, t]

__global__ __launch_bounds__(128) void attn_kernel(const bf16_t* __restrict__ qb,
                                                   const bf16_t* __restrict__ kb,
                                                   const bf16_t* __restrict__ vT,
                                                   bf16_t* __restrict__ yb){
  __shared__ __align__(16) float  Sld[4][16][160];
  __shared__ __align__(16) bf16_t Pld[4][16][160];
  const int lane = threadIdx.x & 31;
  const int w    = threadIdx.x >> 5;
  const int t0   = blockIdx.x * 16;
  const int head = blockIdx.y * 4 + w;
  const int kvh  = head >> 3;                  // GROUP = 8
  const int s_start = (t0 > WINDOW) ? (t0 - WINDOW) : 0;   // multiple of 16
  const int span    = t0 + 16 - s_start;                   // <= 144
  const int ntiles  = span >> 4;                           // <= 9
  const int nc = lane & 15, hi = lane >> 4;

  // Q fragments (reused across all key tiles): D=64 -> two K=32 chunks
  v16bf qa0 = wld_a(qb, NQ, t0, head*DH,      lane);
  v16bf qa1 = wld_a(qb, NQ, t0, head*DH + 32, lane);

  // S = (Q K^T) * scale, scale = 1/sqrt(64) = 0.125
  for (int st = 0; st < ntiles; ++st){
    int s0 = s_start + st*16;
    v8f acc = {};
    acc = wmma_bf16(qa0, wld_b(kb, NKVD, s0, kvh*DH,      lane), acc);
    acc = wmma_bf16(qa1, wld_b(kb, NKVD, s0, kvh*DH + 32, lane), acc);
    #pragma unroll
    for (int r = 0; r < 8; ++r)
      Sld[w][8*hi + r][st*16 + nc] = acc[r] * 0.125f;
  }
  __syncthreads();

  // softmax over exactly the valid window [t-128, t]; zero-pad P elsewhere
  if (lane < 16){
    int t = t0 + lane;
    int jlo = t - WINDOW - s_start; if (jlo < 0) jlo = 0;
    int jhi = t - s_start;
    float mx = -3.0e38f;
    for (int j = jlo; j <= jhi; ++j) mx = fmaxf(mx, Sld[w][lane][j]);
    float sum = 0.0f;
    for (int j = jlo; j <= jhi; ++j){
      float e = __expf(Sld[w][lane][j] - mx);
      Sld[w][lane][j] = e;
      sum += e;
    }
    float inv = 1.0f / sum;
    for (int j = 0; j < 160; ++j){
      float p = (j >= jlo && j <= jhi) ? Sld[w][lane][j]*inv : 0.0f;
      Pld[w][lane][j] = (bf16_t)p;
    }
  }
  __syncthreads();

  // O = P V : iterate K=32 chunks of the (zero-padded) key span
  v8f o[4] = {};
  const int nchunks = (span + 31) >> 5;        // <= 5
  for (int sc = 0; sc < nchunks; ++sc){
    ABu ua;     // A = P fragment 16x32 from LDS
    const bf16_t* pp = &Pld[w][nc][sc*32 + hi*16];
    ua.q[0] = *reinterpret_cast<const uint4*>(pp);
    ua.q[1] = *reinterpret_cast<const uint4*>(pp + 8);
    int sbase = s_start + sc*32;
    #pragma unroll
    for (int dt = 0; dt < 4; ++dt){
      // B = V^T fragment: lane column d = dt*16+nc, 16 contiguous s values
      const bf16_t* vp = vT + ((size_t)kvh*DH + dt*16 + nc)*TP + sbase + hi*16;
      ABu ub;
      ub.q[0] = *reinterpret_cast<const uint4*>(vp);
      ub.q[1] = *reinterpret_cast<const uint4*>(vp + 8);
      o[dt] = wmma_bf16(ua.v, ub.v, o[dt]);
    }
  }
  #pragma unroll
  for (int dt = 0; dt < 4; ++dt){
    #pragma unroll
    for (int r = 0; r < 8; ++r)
      yb[(size_t)(t0 + 8*hi + r)*NQ + head*DH + dt*16 + nc] = (bf16_t)o[dt][r];
  }
}

// ---------------- host ----------------

extern "C" void kernel_launch(void* const* d_in, const int* in_sizes, int n_in,
                              void* d_out, int out_size, void* d_ws, size_t ws_size,
                              hipStream_t stream) {
  const float* x  = (const float*)d_in[0];
  // d_in[1] = positions (arange, implicit in kernels)
  const float* Wq = (const float*)d_in[2];
  const float* bq = (const float*)d_in[3];
  const float* Wk = (const float*)d_in[4];
  const float* bk = (const float*)d_in[5];
  const float* Wv = (const float*)d_in[6];
  const float* bv = (const float*)d_in[7];
  const float* Wo = (const float*)d_in[8];
  const float* bo = (const float*)d_in[9];
  float* out = (float*)d_out;

  char* ws = (char*)d_ws;
  size_t off = 0;
  auto alloc = [&](size_t bytes)->char* {
    char* p = ws + off;
    off += (bytes + 255) & ~(size_t)255;
    return p;
  };
  bf16_t* xb    = (bf16_t*)alloc((size_t)T_SEQ*HID*2);
  bf16_t* wqkvT = (bf16_t*)alloc((size_t)NQKV*HID*2);
  bf16_t* woT   = (bf16_t*)alloc((size_t)HID*NQ*2);
  float*  bqkv  = (float*) alloc((size_t)NQKV*4);
  float*  qkv   = (float*) alloc((size_t)T_SEQ*NQKV*4);
  bf16_t* qb    = (bf16_t*)alloc((size_t)T_SEQ*NQ*2);
  bf16_t* kb    = (bf16_t*)alloc((size_t)T_SEQ*NKVD*2);
  bf16_t* vT    = (bf16_t*)alloc((size_t)NKV_*DH*TP*2);
  bf16_t* yb    = (bf16_t*)alloc((size_t)T_SEQ*NQ*2);
  (void)ws_size; (void)in_sizes; (void)n_in; (void)out_size;

  int n;
  n = T_SEQ*HID;
  f32_to_bf16_kernel<<<(n+255)/256, 256, 0, stream>>>(x, xb, n);
  n = NQKV*HID;
  build_wqkvT_kernel<<<(n+255)/256, 256, 0, stream>>>(Wq, Wk, Wv, wqkvT);
  n = HID*NQ;
  build_woT_kernel<<<(n+255)/256, 256, 0, stream>>>(Wo, woT);
  concat_bias_kernel<<<(NQKV+255)/256, 256, 0, stream>>>(bq, bk, bv, bqkv);

  // fused QKV projection: [T,2880] x [2880,5120] + bias -> f32
  gemm_bf16_kernel<<<dim3(NQKV/64, T_SEQ/256), 128, 0, stream>>>(
      xb, wqkvT, bqkv, qkv, T_SEQ, NQKV, HID);

  n = T_SEQ*(NQ/2);
  rope_q_kernel<<<(n+255)/256, 256, 0, stream>>>(qkv, qb);
  n = T_SEQ*(NKVD/2);
  rope_k_kernel<<<(n+255)/256, 256, 0, stream>>>(qkv, kb);
  n = NKV_*DH*TP;
  vtrans_kernel<<<(n+255)/256, 256, 0, stream>>>(qkv, vT);

  attn_kernel<<<dim3(T_SEQ/16, NH_/4), 128, 0, stream>>>(qb, kb, vT, yb);

  // output projection: [T,4096] x [4096,2880] + bo -> f32 out
  gemm_bf16_kernel<<<dim3(HID/64, T_SEQ/256), 128, 0, stream>>>(
      yb, woT, bo, out, T_SEQ, HID, NQ);
}
